// AttentionDecoder_652835029055
// MI455X (gfx1250) — matrix operand order repaired and checked by reading, not compiled, not run on hardware
//
#include <hip/hip_runtime.h>
#include <hip/hip_bf16.h>
#include <math.h>

// ---------------- problem constants ----------------
#define BB   64
#define NN   196
#define CC   512
#define EE   512
#define HH   1024
#define VV   10000
#define TT   20
#define G4H  4096      // 4*H
#define KG   1024      // E + C  (== H)
#define START_IDX 1

typedef __bf16 bf16;
typedef __attribute__((ext_vector_type(16))) __bf16 bf16x16;
typedef __attribute__((ext_vector_type(8)))  __bf16 bf16x8;
typedef __attribute__((ext_vector_type(8)))  float  f32x8;

// ---------------- WMMA tile helpers (ISA 7.12.2 layouts, wave32) ----------------
// A: 16x32 bf16, row-major source, lda elements per row.
// lanes 0-15: M=lane,  elems = K{k0..k0+7} then K{k0+16..k0+23}
// lanes 16-31: M=lane-16, elems = K{k0+8..k0+15} then K{k0+24..k0+31}
__device__ __forceinline__ bf16x16 load_a_tile(const bf16* A, int lda, int row0, int k0) {
    int lane = threadIdx.x & 31;
    int m    = lane & 15;
    int kb   = k0 + ((lane >> 4) ? 8 : 0);
    const bf16* p = A + (size_t)(row0 + m) * lda + kb;
    bf16x8 lo = *reinterpret_cast<const bf16x8*>(p);        // 16B aligned (kb % 8 == 0)
    bf16x8 hi = *reinterpret_cast<const bf16x8*>(p + 16);
    bf16x16 a;
#pragma unroll
    for (int i = 0; i < 8; ++i) { a[i] = lo[i]; a[i + 8] = hi[i]; }
    return a;
}

// B operand register layout (32x16 tile): lane L<16 -> N=L, elems K=k0..k0+7 then K=k0+16..k0+23;
// lane L>=16 -> N=L-16, elems K=k0+8..k0+15 then K=k0+24..k0+31.
// Weights are PRE-PACKED so each lane's 16 bf16 elements are contiguous:
//   P[((tile_k * ntn + tile_n) * 32 + lane) * 16 + i]
// => two contiguous global_load_b128 per tile, zero packing VALU.
__device__ __forceinline__ bf16x16 load_b_packed(const bf16* P, int ntn, int k0, int col0,
                                                 bool prefetch_next) {
    int lane = threadIdx.x & 31;
    size_t tile = (size_t)(k0 >> 5) * ntn + (col0 >> 4);
    const bf16* p = P + (tile * 32 + lane) * 16;
    bf16x8 lo = *reinterpret_cast<const bf16x8*>(p);
    bf16x8 hi = *reinterpret_cast<const bf16x8*>(p + 8);
    if (prefetch_next)  // next K-tile, same column tile -> global_prefetch_b8
        __builtin_prefetch(p + (size_t)ntn * 512, 0, 1);
    bf16x16 b;
#pragma unroll
    for (int i = 0; i < 8; ++i) { b[i] = lo[i]; b[i + 8] = hi[i]; }
    return b;
}

// C/D f32 16x16: lanes 0-15 -> N=lane, VGPR r = M=r ; lanes 16-31 -> N=lane-16, M=8+r
__device__ __forceinline__ int cd_col(int col0) { return col0 + (threadIdx.x & 15); }
__device__ __forceinline__ int cd_row(int row0, int r) {
    return row0 + (((threadIdx.x & 31) >> 4) ? 8 : 0) + r;
}

__device__ __forceinline__ f32x8 wmma_bf16(bf16x16 a, bf16x16 b, f32x8 c) {
    return __builtin_amdgcn_wmma_f32_16x16x32_bf16(false, a, false, b, (short)0, c,
                                                   false, false);
}

// ---------------- setup kernels ----------------
__global__ void k_f32_to_bf16(const float* __restrict__ src, bf16* __restrict__ dst, int n) {
    int i = blockIdx.x * blockDim.x + threadIdx.x;
    if (i < n) dst[i] = (bf16)src[i];
}

// Convert + pack a row-major [K x N] f32 weight into WMMA-B tiled bf16 layout.
// One thread per (tile, lane): writes 16 contiguous bf16 (32B, fully coalesced).
__global__ void k_pack_b(const float* __restrict__ W, bf16* __restrict__ P, int K, int N) {
    int idx = blockIdx.x * blockDim.x + threadIdx.x;
    int ntn = N >> 4;
    int total = (K >> 5) * ntn * 32;
    if (idx >= total) return;
    int lane = idx & 31;
    int tile = idx >> 5;
    int tk = tile / ntn, tn = tile - tk * ntn;
    int kb = tk * 32 + ((lane >> 4) << 3);
    int n  = tn * 16 + (lane & 15);
    bf16* dst = P + (size_t)idx * 16;
#pragma unroll
    for (int i = 0; i < 16; ++i) dst[i] = (bf16)W[(size_t)(kb + i + ((i >> 3) << 3)) * N + n];
}

// Xm = mean_n X ; ctx = sum_n X (softmax over size-1 axis => weights==1).
// Also drop bf16(ctx) into the right half of the gates A-matrix.
__global__ void k_init_ctx(const float* __restrict__ X, float* __restrict__ Xm,
                           float* __restrict__ ctx, bf16* __restrict__ Ag) {
    int i = blockIdx.x * blockDim.x + threadIdx.x;
    if (i >= BB * CC) return;
    int b = i >> 9, cc = i & (CC - 1);
    const float* xp = X + (size_t)b * NN * CC + cc;
    float s = 0.f;
    for (int n = 0; n < NN; ++n) s += xp[(size_t)n * CC];
    Xm[i]  = s / (float)NN;
    ctx[i] = s;
    Ag[b * KG + EE + cc] = (bf16)s;
}

// out[b,j] = act(sum_k A[b,k] * W[k,j] + bias[j]); one-time f32 path (h0, c0, ctx@Wlc)
__global__ void k_matvec(const float* __restrict__ A, const float* __restrict__ W,
                         const float* __restrict__ bias, float* __restrict__ out,
                         int K, int Ncols, int do_tanh) {
    int i = blockIdx.x * blockDim.x + threadIdx.x;
    if (i >= BB * Ncols) return;
    int b = i / Ncols, j = i - b * Ncols;
    const float* a = A + (size_t)b * K;
    float s = bias[j];
    for (int k = 0; k < K; ++k) s += a[k] * W[(size_t)k * Ncols + j];
    out[i] = do_tanh ? tanhf(s) : s;
}

// out[:,0,:] = one_hot(START_IDX); tok = START_IDX
__global__ void k_out0(float* __restrict__ out, int* __restrict__ tok) {
    int i = blockIdx.x * blockDim.x + threadIdx.x;
    if (i >= BB * VV) return;
    int b = i / VV, v = i - b * VV;
    out[(size_t)b * TT * VV + v] = (v == START_IDX) ? 1.f : 0.f;
    if (v == 0) tok[b] = START_IDX;
}

// e = emb[tok]; f32 copy for comb, bf16 into left half of gates A-matrix
__global__ void k_gather_e(const float* __restrict__ emb, const int* __restrict__ tok,
                           float* __restrict__ e, bf16* __restrict__ Ag) {
    int i = blockIdx.x * blockDim.x + threadIdx.x;
    if (i >= BB * EE) return;
    int b = i >> 9, j = i & (EE - 1);
    float ev = emb[(size_t)tok[b] * EE + j];
    e[i] = ev;
    Ag[b * KG + j] = (bf16)ev;
}

// ---------------- WMMA GEMM kernels (1 wave per 16x16 tile) ----------------
// gates[64,4096] = Ag[64,1024]@Wih + hb[64,1024]@Whh + b_ih + b_hh
__global__ void __launch_bounds__(32)
k_gates(const bf16* __restrict__ Ag, const bf16* __restrict__ hb,
        const bf16* __restrict__ WihP, const bf16* __restrict__ WhhP,
        const float* __restrict__ bih, const float* __restrict__ bhh,
        float* __restrict__ gates) {
    int col0 = blockIdx.x * 16, row0 = blockIdx.y * 16;
    const int ntn = G4H >> 4;  // 256
    f32x8 acc = {};
    for (int k0 = 0; k0 < KG; k0 += 32) {
        bool pf = (k0 + 32) < KG;
        acc = wmma_bf16(load_a_tile(Ag, KG, row0, k0),
                        load_b_packed(WihP, ntn, k0, col0, pf), acc);
        acc = wmma_bf16(load_a_tile(hb, HH, row0, k0),
                        load_b_packed(WhhP, ntn, k0, col0, pf), acc);
    }
    int col = cd_col(col0);
    float bias = bih[col] + bhh[col];
#pragma unroll
    for (int r = 0; r < 8; ++r)
        gates[(size_t)cd_row(row0, r) * G4H + col] = acc[r] + bias;
}

// torch LSTMCell gate order: i, f, g, o. Writes f32 h AND bf16 hb (for next GEMMs).
__global__ void k_cell(const float* __restrict__ gates, float* __restrict__ h,
                       float* __restrict__ c, bf16* __restrict__ hb) {
    int i = blockIdx.x * blockDim.x + threadIdx.x;
    if (i >= BB * HH) return;
    int b = i >> 10, j = i & (HH - 1);
    const float* g = gates + (size_t)b * G4H;
    float gi = g[j], gf = g[HH + j], gg = g[2 * HH + j], go = g[3 * HH + j];
    float si = 1.f / (1.f + expf(-gi));
    float sf = 1.f / (1.f + expf(-gf));
    float so = 1.f / (1.f + expf(-go));
    float c2 = sf * c[i] + si * tanhf(gg);
    float h2 = so * tanhf(c2);
    c[i] = c2;
    h[i] = h2;
    hb[i] = (bf16)h2;
}

// combB[64,512] = bf16( h2@Wlh + blh + e + ctxWlc )   (ctxWlc already includes blc)
__global__ void __launch_bounds__(32)
k_comb(const bf16* __restrict__ hb, const bf16* __restrict__ WlhP,
       const float* __restrict__ blh, const float* __restrict__ e,
       const float* __restrict__ ctxWlc, bf16* __restrict__ combB) {
    int col0 = blockIdx.x * 16, row0 = blockIdx.y * 16;
    const int ntn = EE >> 4;  // 32
    f32x8 acc = {};
    for (int k0 = 0; k0 < HH; k0 += 32)
        acc = wmma_bf16(load_a_tile(hb, HH, row0, k0),
                        load_b_packed(WlhP, ntn, k0, col0, (k0 + 32) < HH), acc);
    int col = cd_col(col0);
#pragma unroll
    for (int r = 0; r < 8; ++r) {
        int row = cd_row(row0, r);
        float v = acc[r] + blh[col] + e[(size_t)row * EE + col] + ctxWlc[(size_t)row * EE + col];
        combB[(size_t)row * EE + col] = (bf16)v;
    }
}

// Z[64,10000] = combB[64,512] @ Wout + bout
__global__ void __launch_bounds__(32)
k_logits(const bf16* __restrict__ combB, const bf16* __restrict__ WoutP,
         const float* __restrict__ bout, float* __restrict__ Z) {
    int col0 = blockIdx.x * 16, row0 = blockIdx.y * 16;
    const int ntn = VV >> 4;  // 625
    f32x8 acc = {};
    for (int k0 = 0; k0 < EE; k0 += 32)
        acc = wmma_bf16(load_a_tile(combB, EE, row0, k0),
                        load_b_packed(WoutP, ntn, k0, col0, (k0 + 32) < EE), acc);
    int col = cd_col(col0);
    float bias = bout[col];
#pragma unroll
    for (int r = 0; r < 8; ++r)
        Z[(size_t)cd_row(row0, r) * VV + col] = acc[r] + bias;
}

// Per batch row: log_softmax into out[:,t,:] and argmax -> tok (first-index tie-break)
__global__ void __launch_bounds__(256)
k_logsm(const float* __restrict__ Z, float* __restrict__ out, int* __restrict__ tok, int t) {
    __shared__ float sv[256];
    __shared__ int   si[256];
    __shared__ float ss[256];
    int b = blockIdx.x, tid = threadIdx.x;
    const float* z = Z + (size_t)b * VV;
    float best = -3.4e38f; int bidx = 0;
    for (int v = tid; v < VV; v += 256) {
        float x = z[v];
        if (x > best) { best = x; bidx = v; }
    }
    sv[tid] = best; si[tid] = bidx;
    __syncthreads();
    for (int s = 128; s > 0; s >>= 1) {
        if (tid < s) {
            float ov = sv[tid + s]; int oi = si[tid + s];
            if (ov > sv[tid] || (ov == sv[tid] && oi < si[tid])) { sv[tid] = ov; si[tid] = oi; }
        }
        __syncthreads();
    }
    float gmax = sv[0]; int gidx = si[0];
    float lsum = 0.f;
    for (int v = tid; v < VV; v += 256) lsum += expf(z[v] - gmax);
    ss[tid] = lsum;
    __syncthreads();
    for (int s = 128; s > 0; s >>= 1) {
        if (tid < s) ss[tid] += ss[tid + s];
        __syncthreads();
    }
    float logZ = gmax + logf(ss[0]);
    float* o = out + (size_t)b * TT * VV + (size_t)t * VV;
    for (int v = tid; v < VV; v += 256) o[v] = z[v] - logZ;
    if (tid == 0) tok[b] = gidx;
}

// ---------------- host launcher ----------------
extern "C" void kernel_launch(void* const* d_in, const int* in_sizes, int n_in,
                              void* d_out, int out_size, void* d_ws, size_t ws_size,
                              hipStream_t stream) {
    (void)in_sizes; (void)n_in; (void)out_size; (void)ws_size;
    const float* X    = (const float*)d_in[0];
    const float* emb  = (const float*)d_in[1];
    const float* W_ih = (const float*)d_in[2];
    const float* b_ih = (const float*)d_in[3];
    const float* W_hh = (const float*)d_in[4];
    const float* b_hh = (const float*)d_in[5];
    // d_in[6..11] (Wa,ba,Wh,bh,Wo,bo): dead code — softmax over size-1 axis => weights==1
    const float* Wlh  = (const float*)d_in[12];
    const float* blh  = (const float*)d_in[13];
    const float* Wlc  = (const float*)d_in[14];
    const float* blc  = (const float*)d_in[15];
    const float* Wout = (const float*)d_in[16];
    const float* bout = (const float*)d_in[17];
    const float* Wc0  = (const float*)d_in[18];
    const float* bc0  = (const float*)d_in[19];
    const float* Wh0  = (const float*)d_in[20];
    const float* bh0  = (const float*)d_in[21];
    float* out = (float*)d_out;

    char* ws = (char*)d_ws;
    size_t off = 0;
    auto alloc = [&](size_t bytes) -> void* {
        void* p = ws + off;
        off = (off + bytes + 255) & ~(size_t)255;
        return p;
    };
    bf16*  WihP   = (bf16*)alloc((size_t)KG * G4H * 2);   // packed WMMA-B layout, 8 MB
    bf16*  WhhP   = (bf16*)alloc((size_t)HH * G4H * 2);   // 8 MB
    bf16*  WlhP   = (bf16*)alloc((size_t)HH * EE * 2);    // 1 MB
    bf16*  WoutP  = (bf16*)alloc((size_t)EE * VV * 2);    // 10 MB
    bf16*  Ag     = (bf16*)alloc((size_t)BB * KG * 2);    // [e | ctx] bf16
    bf16*  hb     = (bf16*)alloc((size_t)BB * HH * 2);
    bf16*  combB  = (bf16*)alloc((size_t)BB * EE * 2);
    float* Xm     = (float*)alloc((size_t)BB * CC * 4);
    float* ctx    = (float*)alloc((size_t)BB * CC * 4);
    float* ctxWlc = (float*)alloc((size_t)BB * EE * 4);
    float* e      = (float*)alloc((size_t)BB * EE * 4);
    float* h      = (float*)alloc((size_t)BB * HH * 4);
    float* c      = (float*)alloc((size_t)BB * HH * 4);
    float* gates  = (float*)alloc((size_t)BB * G4H * 4);
    float* Z      = (float*)alloc((size_t)BB * VV * 4);
    int*   tok    = (int*)alloc(BB * sizeof(int));

    const int TPB = 256;
    auto blocks = [](int n, int t) { return (n + t - 1) / t; };

    // --- one-time (per call): pack weights to WMMA-B bf16 tiles; ctx/Xm; h0/c0; ctx@Wlc
    k_pack_b<<<blocks(KG * G4H / 16, TPB), TPB, 0, stream>>>(W_ih, WihP, KG, G4H);
    k_pack_b<<<blocks(HH * G4H / 16, TPB), TPB, 0, stream>>>(W_hh, WhhP, HH, G4H);
    k_pack_b<<<blocks(HH * EE / 16, TPB), TPB, 0, stream>>>(Wlh, WlhP, HH, EE);
    k_pack_b<<<blocks(EE * VV / 16, TPB), TPB, 0, stream>>>(Wout, WoutP, EE, VV);

    k_init_ctx<<<blocks(BB * CC, TPB), TPB, 0, stream>>>(X, Xm, ctx, Ag);
    k_matvec<<<blocks(BB * HH, TPB), TPB, 0, stream>>>(Xm, Wh0, bh0, h, CC, HH, 1);
    k_matvec<<<blocks(BB * HH, TPB), TPB, 0, stream>>>(Xm, Wc0, bc0, c, CC, HH, 1);
    k_matvec<<<blocks(BB * EE, TPB), TPB, 0, stream>>>(ctx, Wlc, blc, ctxWlc, CC, EE, 0);
    k_f32_to_bf16<<<blocks(BB * HH, TPB), TPB, 0, stream>>>(h, hb, BB * HH);
    k_out0<<<blocks(BB * VV, TPB), TPB, 0, stream>>>(out, tok);

    // --- 19 sequential decoder steps
    for (int t = 1; t < TT; ++t) {
        k_gather_e<<<blocks(BB * EE, TPB), TPB, 0, stream>>>(emb, tok, e, Ag);
        k_gates<<<dim3(G4H / 16, BB / 16), 32, 0, stream>>>(Ag, hb, WihP, WhhP,
                                                            b_ih, b_hh, gates);
        k_cell<<<blocks(BB * HH, TPB), TPB, 0, stream>>>(gates, h, c, hb);
        k_comb<<<dim3(EE / 16, BB / 16), 32, 0, stream>>>(hb, WlhP, blh, e, ctxWlc, combB);
        k_logits<<<dim3(VV / 16, BB / 16), 32, 0, stream>>>(combB, WoutP, bout, Z);
        k_logsm<<<BB, 256, 0, stream>>>(Z, out, tok, t);
    }
}
